// DyGrEncoder_66425964200050
// MI455X (gfx1250) — matrix operand order repaired and checked by reading, not compiled, not run on hardware
//
#include <hip/hip_runtime.h>
#include <stdint.h>
#include <stddef.h>

typedef __attribute__((ext_vector_type(16))) __bf16 v16bf;
typedef __attribute__((ext_vector_type(8)))  float  v8f;

static constexpr int Dk = 128;   // feature dim = K of every GEMM
static constexpr int SP = 136;   // padded LDS row stride (bf16 elems) -> conflict-free A reads

// ---- WMMA fragment helpers (wave32, 16x16x32 bf16) -------------------------
// A (16x32, MxK) per-lane layout: g=lane/16, m=lane%16; elems e0..15 -> K=g*16+e
// (contiguous, matching glbB)
__device__ __forceinline__ v16bf ldsA(const __bf16* As, int kb, int lane) {
  const int g = lane >> 4, m = lane & 15;
  const __bf16* row = As + m * SP + kb;
  v16bf a;
#pragma unroll
  for (int j = 0; j < 16; ++j) a[j] = row[g * 16 + j];
  return a;
}

// B (32x16, KxN): lanes 0-15 hold K=0..15 for col n=lane, lanes 16-31 hold K=16..31.
// Weight stored row-major [out][128] bf16, so B[k][nb+n] = W[nb+n][k] -> 16 contiguous
// bf16 (32B) per lane.
__device__ __forceinline__ v16bf glbB(const __bf16* W, int nb, int kb, int lane) {
  const int g = lane >> 4, n = lane & 15;
  const __bf16* p = W + (size_t)(nb + n) * Dk + kb + g * 16;
  v16bf b;
#pragma unroll
  for (int j = 0; j < 16; ++j) b[j] = p[j];
  return b;
}

__device__ __forceinline__ v8f wmma_bf16(v16bf a, v16bf b, v8f c) {
  return __builtin_amdgcn_wmma_f32_16x16x32_bf16(false, a, false, b, (short)0, c,
                                                 false, false);
}

// Guaranteed-native fp32 atomic add (no return -> STOREcnt; drained by S_ENDPGM's
// implicit wait-idle). Device scope so cross-WGP accumulation is coherent at L2.
__device__ __forceinline__ void atomAddF32(float* p, float v) {
  asm volatile("global_atomic_add_f32 %0, %1, off scope:SCOPE_DEV"
               :: "v"(p), "v"(v) : "memory");
}

__device__ __forceinline__ float sigf(float x) { return 1.0f / (1.0f + __expf(-x)); }

// Stage a 16-row fp32 tile into LDS as bf16 (coalesced float4 reads).
__device__ __forceinline__ void stage16(const float* __restrict__ src, __bf16* dst,
                                        int row0, int N, int tid) {
  for (int i = tid; i < 16 * 32; i += 256) {
    int r = i >> 5, c = (i & 31) << 2;
    int gr = row0 + r; if (gr >= N) gr = N - 1;   // clamp; WMMA needs full EXEC
    float4 v = reinterpret_cast<const float4*>(src + (size_t)gr * Dk)[c >> 2];
    __bf16* d = dst + r * SP + c;
    d[0] = (__bf16)v.x; d[1] = (__bf16)v.y; d[2] = (__bf16)v.z; d[3] = (__bf16)v.w;
  }
}

// ---- one-time weight conversion (bf16; conv_W transposed to [out][in]) -----
__global__ __launch_bounds__(256) void prep_weights(
    const float* __restrict__ convW, const float* __restrict__ gWih,
    const float* __restrict__ gWhh, const float* __restrict__ lWih,
    const float* __restrict__ lWhh,
    __bf16* __restrict__ convWt, __bf16* __restrict__ gWihB,
    __bf16* __restrict__ gWhhB, __bf16* __restrict__ lWihB,
    __bf16* __restrict__ lWhhB) {
  int t = blockIdx.x * 256 + threadIdx.x;
  if (t < 3 * Dk * Dk) {           // conv_W [3][k][j] used as x@W -> store [3][j][k]
    int l = t >> 14, rem = t & 16383, k = rem >> 7, j = rem & 127;
    convWt[(l << 14) + j * Dk + k] = (__bf16)convW[t];
  }
  if (t < 3 * Dk * Dk) { gWihB[t] = (__bf16)gWih[t]; gWhhB[t] = (__bf16)gWhh[t]; }
  if (t < 4 * Dk * Dk) { lWihB[t] = (__bf16)lWih[t]; lWhhB[t] = (__bf16)lWhh[t]; }
}

// ---- m = h @ W  (N x 128 x 128), output bf16 for cheap edge gather ---------
__global__ __launch_bounds__(256) void conv_gemm(
    const float* __restrict__ h, const __bf16* __restrict__ Wt,
    __bf16* __restrict__ m, int N) {
  __shared__ __bf16 As[128 * SP];
  const int row0 = blockIdx.x * 128;
  const int tid = threadIdx.x;
  for (int i = tid; i < 128 * 32; i += 256) {
    int r = i >> 5, c = (i & 31) << 2;
    int gr = row0 + r; if (gr >= N) gr = N - 1;
    float4 v = reinterpret_cast<const float4*>(h + (size_t)gr * Dk)[c >> 2];
    __bf16* d = As + r * SP + c;
    d[0] = (__bf16)v.x; d[1] = (__bf16)v.y; d[2] = (__bf16)v.z; d[3] = (__bf16)v.w;
  }
  __syncthreads();
  const int wave = tid >> 5, lane = tid & 31;
  const __bf16* Aw = As + wave * 16 * SP;       // each wave owns 16 rows
  v8f zero = {};
  v8f acc[8];
#pragma unroll
  for (int i = 0; i < 8; ++i) acc[i] = zero;
#pragma unroll
  for (int ks = 0; ks < 4; ++ks) {
    v16bf a = ldsA(Aw, ks * 32, lane);          // A frag reused across 8 N-tiles
#pragma unroll
    for (int nc = 0; nc < 8; ++nc)
      acc[nc] = wmma_bf16(a, glbB(Wt, nc * 16, ks * 32, lane), acc[nc]);
  }
  // Transpose accumulators through LDS (each wave writes only its own 16 rows,
  // and has finished reading them), then do fully coalesced b128 stores.
  const int rr = (lane >> 4) * 8, cc = lane & 15;
  const int rbase = wave * 16 + rr;
#pragma unroll
  for (int nc = 0; nc < 8; ++nc)
#pragma unroll
    for (int v = 0; v < 8; ++v)
      As[(rbase + v) * SP + nc * 16 + cc] = (__bf16)acc[nc][v];
  __syncthreads();
  for (int i = tid; i < 128 * 16; i += 256) {   // 128 rows x 16 chunks of 16B
    int r = i >> 4, c8 = (i & 15) << 3;
    int gr = row0 + r;
    if (gr < N) {
      uint4 vv = *reinterpret_cast<const uint4*>(As + r * SP + c8);
      *reinterpret_cast<uint4*>(m + (size_t)gr * Dk + c8) = vv;
    }
  }
}

// ---- agg[dst] += m[src] * w  (one wave per edge) ---------------------------
// Address pattern: per atomic instruction, 32 lanes cover one full 128B line.
__global__ __launch_bounds__(256) void scatter_add(
    const __bf16* __restrict__ m, const int* __restrict__ src,
    const int* __restrict__ dst, const float* __restrict__ w,
    float* __restrict__ agg, int E) {
  int t = blockIdx.x * 256 + threadIdx.x;
  int e = t >> 5;
  if (e >= E) return;
  int lane = t & 31;
  int s = src[e], d = dst[e];
  float ww = w[e];
  const __bf16* mp = m + (size_t)s * Dk + lane;
  float* ap = agg + (size_t)d * Dk + lane;
#pragma unroll
  for (int j = 0; j < 4; ++j)
    atomAddF32(ap + j * 32, (float)mp[j * 32] * ww);
}

// ---- fused GRU cell: gi = agg@Wih^T+bih, gh = h@Whh^T+bhh, gates, h' -------
__global__ __launch_bounds__(256) void gru_fused(
    const float* __restrict__ agg, const float* __restrict__ h,
    const __bf16* __restrict__ Wih, const __bf16* __restrict__ Whh,
    const float* __restrict__ bih, const float* __restrict__ bhh,
    float* __restrict__ hout, int N) {
  __shared__ __bf16 Aa[16 * SP];
  __shared__ __bf16 Ah[16 * SP];
  __shared__ float gi[16][384];
  __shared__ float gh[16][384];
  const int row0 = blockIdx.x * 16;
  const int tid = threadIdx.x;
  stage16(agg, Aa, row0, N, tid);
  stage16(h,   Ah, row0, N, tid);
  __syncthreads();
  const int wave = tid >> 5, lane = tid & 31;
  const int rr = (lane >> 4) * 8, cc = lane & 15;
  for (int t = wave; t < 48; t += 8) {          // 24 gi tiles + 24 gh tiles
    const bool isI = (t < 24);
    const int c = isI ? t : t - 24;
    const __bf16* A = isI ? Aa : Ah;
    const __bf16* W = isI ? Wih : Whh;
    v8f acc = {};
#pragma unroll
    for (int ks = 0; ks < 4; ++ks)
      acc = wmma_bf16(ldsA(A, ks * 32, lane), glbB(W, c * 16, ks * 32, lane), acc);
    const float bias = (isI ? bih : bhh)[c * 16 + cc];
    float (*G)[384] = isI ? gi : gh;
#pragma unroll
    for (int v = 0; v < 8; ++v) G[rr + v][c * 16 + cc] = acc[v] + bias;
  }
  __syncthreads();
  for (int i = tid; i < 16 * Dk; i += 256) {
    int r = i >> 7, c = i & 127;
    int gr = row0 + r;
    if (gr >= N) continue;
    float rg = sigf(gi[r][c] + gh[r][c]);
    float zg = sigf(gi[r][128 + c] + gh[r][128 + c]);
    float ng = tanhf(gi[r][256 + c] + rg * gh[r][256 + c]);
    float hv = h[(size_t)gr * Dk + c];
    hout[(size_t)gr * Dk + c] = (1.0f - zg) * ng + zg * hv;
  }
}

// ---- fused single-step LSTM; writes (H_new, H_new, C_new) ------------------
__global__ __launch_bounds__(256) void lstm_fused(
    const float* __restrict__ x, const float* __restrict__ H,
    const float* __restrict__ C,
    const __bf16* __restrict__ Wih, const __bf16* __restrict__ Whh,
    const float* __restrict__ bih, const float* __restrict__ bhh,
    float* __restrict__ out, int N) {
  __shared__ __bf16 Ax[16 * SP];
  __shared__ __bf16 Ah[16 * SP];
  __shared__ float gates[16][512];
  const int row0 = blockIdx.x * 16;
  const int tid = threadIdx.x;
  stage16(x, Ax, row0, N, tid);
  stage16(H, Ah, row0, N, tid);
  __syncthreads();
  const int wave = tid >> 5, lane = tid & 31;
  const int rr = (lane >> 4) * 8, cc = lane & 15;
  for (int c = wave; c < 32; c += 8) {          // both GEMMs share accumulator
    v8f acc = {};
#pragma unroll
    for (int ks = 0; ks < 4; ++ks)
      acc = wmma_bf16(ldsA(Ax, ks * 32, lane), glbB(Wih, c * 16, ks * 32, lane), acc);
#pragma unroll
    for (int ks = 0; ks < 4; ++ks)
      acc = wmma_bf16(ldsA(Ah, ks * 32, lane), glbB(Whh, c * 16, ks * 32, lane), acc);
    const float bias = bih[c * 16 + cc] + bhh[c * 16 + cc];
#pragma unroll
    for (int v = 0; v < 8; ++v) gates[rr + v][c * 16 + cc] = acc[v] + bias;
  }
  __syncthreads();
  const size_t NH = (size_t)N * Dk;
  for (int i = tid; i < 16 * Dk; i += 256) {
    int r = i >> 7, c = i & 127;
    int gr = row0 + r;
    if (gr >= N) continue;
    float ig = sigf(gates[r][c]);
    float fg = sigf(gates[r][128 + c]);
    float gg = tanhf(gates[r][256 + c]);
    float og = sigf(gates[r][384 + c]);
    float cn = fg * C[(size_t)gr * Dk + c] + ig * gg;
    float hn = og * tanhf(cn);
    out[(size_t)gr * Dk + c] = hn;
    out[NH + (size_t)gr * Dk + c] = hn;
    out[2 * NH + (size_t)gr * Dk + c] = cn;
  }
}

extern "C" void kernel_launch(void* const* d_in, const int* in_sizes, int n_in,
                              void* d_out, int out_size, void* d_ws, size_t ws_size,
                              hipStream_t stream) {
  const float* X     = (const float*)d_in[0];
  const int*   ei    = (const int*)d_in[1];
  const float* ew    = (const float*)d_in[2];
  const float* Hh    = (const float*)d_in[3];
  const float* Cc    = (const float*)d_in[4];
  const float* convW = (const float*)d_in[5];
  const float* gWih  = (const float*)d_in[6];
  const float* gWhh  = (const float*)d_in[7];
  const float* gbih  = (const float*)d_in[8];
  const float* gbhh  = (const float*)d_in[9];
  const float* lWih  = (const float*)d_in[10];
  const float* lWhh  = (const float*)d_in[11];
  const float* lbih  = (const float*)d_in[12];
  const float* lbhh  = (const float*)d_in[13];
  (void)n_in; (void)out_size; (void)ws_size;

  const int N = in_sizes[0] / Dk;
  const int E = in_sizes[2];
  const int* src = ei;        // edge_index[0]
  const int* dst = ei + E;    // edge_index[1]

  char* ws = (char*)d_ws;
  size_t off = 0;
  auto take = [&](size_t bytes) {
    char* p = ws + off;
    off = (off + bytes + 255) & ~(size_t)255;
    return p;
  };
  float*  hA     = (float*)take((size_t)N * Dk * 4);
  float*  hB     = (float*)take((size_t)N * Dk * 4);
  float*  agg    = (float*)take((size_t)N * Dk * 4);
  __bf16* m      = (__bf16*)take((size_t)N * Dk * 2);
  __bf16* convWt = (__bf16*)take((size_t)3 * Dk * Dk * 2);
  __bf16* gWihB  = (__bf16*)take((size_t)3 * Dk * Dk * 2);
  __bf16* gWhhB  = (__bf16*)take((size_t)3 * Dk * Dk * 2);
  __bf16* lWihB  = (__bf16*)take((size_t)4 * Dk * Dk * 2);
  __bf16* lWhhB  = (__bf16*)take((size_t)4 * Dk * Dk * 2);

  prep_weights<<<256, 256, 0, stream>>>(convW, gWih, gWhh, lWih, lWhh,
                                        convWt, gWihB, gWhhB, lWihB, lWhhB);

  const float* hcur = X;
  float* hnext = hA;
  for (int l = 0; l < 3; ++l) {
    conv_gemm<<<(N + 127) / 128, 256, 0, stream>>>(
        hcur, convWt + (size_t)l * Dk * Dk, m, N);
    hipMemsetAsync(agg, 0, (size_t)N * Dk * 4, stream);
    scatter_add<<<(int)(((size_t)E * 32 + 255) / 256), 256, 0, stream>>>(
        m, src, dst, ew, agg, E);
    gru_fused<<<(N + 15) / 16, 256, 0, stream>>>(
        agg, hcur, gWihB, gWhhB, gbih, gbhh, hnext, N);
    hcur = hnext;
    hnext = (hnext == hA) ? hB : hA;
  }
  lstm_fused<<<(N + 15) / 16, 256, 0, stream>>>(
      hcur, Hh, Cc, lWihB, lWhhB, lbih, lbhh, (float*)d_out, N);
}
